// ControlPointDecoder_17858474017193
// MI455X (gfx1250) — matrix operand order repaired and checked by reading, not compiled, not run on hardware
//
#include <hip/hip_runtime.h>

// ---------------------------------------------------------------------------
// ControlPointDecoder for MI455X (gfx1250, wave32, WMMA bf16)
//
// - All GEMMs: v_wmma_f32_16x16x32_bf16, fp32 accumulate, bf16 intermediates.
// - Weights pre-transposed to N-major bf16 (zero-padded to 128-row multiples)
//   so A- and B-tile staging are identical b128 copies, no scatter/guards.
// - LDS staging via GLOBAL_LOAD_ASYNC_TO_LDS_B128 (ASYNCcnt) with double
//   buffering: next K-chunk streams into LDS while WMMAs consume current one.
// - latent @ Wq1[0:1024] computed once per batch row (shared across 16 paths),
//   then a rank-3 per-path correction.
// ---------------------------------------------------------------------------

typedef __attribute__((ext_vector_type(16))) __bf16 v16bf;
typedef __attribute__((ext_vector_type(8)))  float  v8f;

#define LATENT_DIM 1024
#define BATCH      4096
#define NPATHS     16
#define NROWS      (BATCH * NPATHS)   // 65536
#define NPOINTS    256

// ---- workspace layout (bytes, all 256-aligned) ----
#define OFF_LATB   ((size_t)0)              // 4096x1024 bf16          8388608
#define OFF_WP1T   ((size_t)8388608)        // 512x1024  bf16 (Wp1^T)  1048576
#define OFF_WP2T   ((size_t)9437184)        // 512x512   bf16 (Wp2^T)   524288
#define OFF_WP3T   ((size_t)9961472)        // 128x512   bf16 (Wp3^T pad)131072
#define OFF_WQ1T   ((size_t)10092544)       // 256x1024  bf16 (Wq1a^T)  524288
#define OFF_WQ2T   ((size_t)10616832)       // 256x256   bf16 (Wq2^T)   131072
#define OFF_WQ3T   ((size_t)10747904)       // 512x256   bf16 (Wq3^T)   262144
#define OFF_H1     ((size_t)11010048)       // 4096x512  bf16          4194304
#define OFF_H2     ((size_t)15204352)       // 4096x512  bf16          4194304
#define OFF_PP     ((size_t)19398656)       // 4096x48   f32            786432
#define OFF_G0     ((size_t)20185088)       // 4096x256  f32           4194304
#define OFF_G1     ((size_t)24379392)       // 65536x256 bf16         33554432
#define OFF_G2     ((size_t)57933824)       // 65536x256 bf16         33554432
// total ~ 87.3 MiB

__device__ __forceinline__ unsigned short f2bf(float f) {
  union { float f; unsigned int u; } v; v.f = f;
  unsigned int u = v.u;
  unsigned int r = u + 0x7FFFu + ((u >> 16) & 1u);   // round-to-nearest-even
  return (unsigned short)(r >> 16);
}

// gfx1250 async DMA: global -> LDS, 16 bytes, tracked by ASYNCcnt.
__device__ __forceinline__ void async_load_b128(const unsigned short* gptr,
                                                unsigned short* lptr) {
  unsigned lds_off = (unsigned)(unsigned long long)(size_t)lptr; // addr[31:0] = LDS offset
  unsigned long long gaddr = (unsigned long long)(size_t)gptr;
  asm volatile("global_load_async_to_lds_b128 %0, %1, off"
               :: "v"(lds_off), "v"(gaddr) : "memory");
}
__device__ __forceinline__ void wait_asynccnt0() {
  asm volatile("s_wait_asynccnt 0x0" ::: "memory");
}

__global__ void conv_f32_bf16(const float* __restrict__ src,
                              unsigned short* __restrict__ dst, int n) {
  for (int i = blockIdx.x * blockDim.x + threadIdx.x; i < n;
       i += gridDim.x * blockDim.x)
    dst[i] = f2bf(src[i]);
}

// Wt[n*K + k] = bf16(W[k*N + n]) for n < N, zero rows for N <= n < Npad.
__global__ void transpose_bf16(const float* __restrict__ W,
                               unsigned short* __restrict__ Wt,
                               int K, int N, int Npad) {
  int idx = blockIdx.x * blockDim.x + threadIdx.x;
  if (idx >= Npad * K) return;
  int n = idx / K, k = idx - n * K;
  float v = (n < N) ? W[(size_t)k * N + n] : 0.0f;
  Wt[idx] = f2bf(v);
}

// ---------------------------------------------------------------------------
// bf16 WMMA GEMM: C(MxN) = act(A(MxK) @ B(KxN) + bias)
// A: row-major bf16 (M x K).  Bt: B transposed, row-major bf16 (Npad x K),
// Npad = gridDim.x*128 (zero-padded).  act: 0=none, 1=relu, 2=sigmoid.
// Block tile 128x128, 8 waves of 32x64, K in chunks of 32, double-buffered
// async LDS staging. Requires M % 128 == 0, K % 32 == 0.
// ---------------------------------------------------------------------------
__global__ __launch_bounds__(256)
void gemm_bf16_wmma(const unsigned short* __restrict__ A,
                    const unsigned short* __restrict__ Bt,
                    const float* __restrict__ bias,
                    float* __restrict__ Cf,            // f32 out (or null)
                    unsigned short* __restrict__ Cb,   // bf16 out (or null)
                    int M, int N, int K, int act) {
  __shared__ __align__(16) unsigned short As[2][128 * 32];  // [m][k]
  __shared__ __align__(16) unsigned short Bs[2][128 * 32];  // [n][k]

  const int tid  = threadIdx.x;
  const int lane = tid & 31;
  const int wave = tid >> 5;
  const int wm   = wave >> 1;      // 0..3 -> 32-row group
  const int wn   = wave & 1;       // 0..1 -> 64-col group
  const int lm   = lane & 15;
  const int lh   = lane >> 4;      // 0/1
  const int m0   = blockIdx.y * 128;
  const int n0   = blockIdx.x * 128;

  v8f acc[2][4];
#pragma unroll
  for (int mi = 0; mi < 2; ++mi)
#pragma unroll
    for (int ni = 0; ni < 4; ++ni) acc[mi][ni] = (v8f){0,0,0,0,0,0,0,0};

  // Stage one K-chunk (32) of both tiles into LDS buffer `buf`.
  auto stage = [&](int buf, int kk) {
#pragma unroll
    for (int i = tid; i < 512; i += 256) {
      int row = i >> 2, seg = (i & 3) * 8;                 // 8 bf16 = 16B
      async_load_b128(A  + (size_t)(m0 + row) * K + kk + seg,
                      &As[buf][row * 32 + seg]);
      async_load_b128(Bt + (size_t)(n0 + row) * K + kk + seg,
                      &Bs[buf][row * 32 + seg]);
    }
  };

  const int nk = K >> 5;
  stage(0, 0);
  for (int c = 0; c < nk; ++c) {
    const int buf = c & 1;
    wait_asynccnt0();      // my share of buf is staged
    __syncthreads();       // everyone's share is staged; prev reads are done
    if (c + 1 < nk) stage(buf ^ 1, (c + 1) << 5);

    // A fragment (16x32): lane holds row m=lm; lanes0-15 K={0..7,16..23},
    // lanes16-31 shifted by 8. Two ds_load_b128 per fragment.
    v16bf af[2];
#pragma unroll
    for (int mi = 0; mi < 2; ++mi) {
      int row = wm * 32 + mi * 16 + lm;
      int kb  = lh * 8;
      v16bf t;
      *reinterpret_cast<uint4*>(&t) =
          *reinterpret_cast<const uint4*>(&As[buf][row * 32 + kb]);
      *(reinterpret_cast<uint4*>(&t) + 1) =
          *reinterpret_cast<const uint4*>(&As[buf][row * 32 + 16 + kb]);
      af[mi] = t;
    }
    // B fragment (32x16): lane holds col n=lm; lanes0-15 K={0..7,16..23},
    // lanes16-31 shifted by 8 (matches A). Bs is n-major.
    v16bf bfr[4];
#pragma unroll
    for (int ni = 0; ni < 4; ++ni) {
      int col = wn * 64 + ni * 16 + lm;
      int kb  = lh * 8;
      v16bf t;
      *reinterpret_cast<uint4*>(&t) =
          *reinterpret_cast<const uint4*>(&Bs[buf][col * 32 + kb]);
      *(reinterpret_cast<uint4*>(&t) + 1) =
          *reinterpret_cast<const uint4*>(&Bs[buf][col * 32 + 16 + kb]);
      bfr[ni] = t;
    }
#pragma unroll
    for (int mi = 0; mi < 2; ++mi)
#pragma unroll
      for (int ni = 0; ni < 4; ++ni)
        acc[mi][ni] = __builtin_amdgcn_wmma_f32_16x16x32_bf16(
            false, af[mi], false, bfr[ni], (short)0, acc[mi][ni], false, false);
    __syncthreads();       // all fragment reads of `buf` done before restage
  }

  // ---- epilogue: bias + activation + store ----
#pragma unroll
  for (int mi = 0; mi < 2; ++mi)
#pragma unroll
    for (int ni = 0; ni < 4; ++ni) {
      int col = n0 + wn * 64 + ni * 16 + lm;
      if (col >= N) continue;
      float bv = bias ? bias[col] : 0.0f;
      int rowbase = m0 + wm * 32 + mi * 16 + lh * 8;
#pragma unroll
      for (int r = 0; r < 8; ++r) {
        float v = acc[mi][ni][r] + bv;
        if (act == 1)      v = fmaxf(v, 0.0f);
        else if (act == 2) v = 1.0f / (1.0f + __expf(-v));
        size_t idx = (size_t)(rowbase + r) * N + col;
        if (Cf) Cf[idx] = v;
        else    Cb[idx] = f2bf(v);
      }
    }
}

// g1[b,p,:] = relu(G0[b,:] + pp[b,p,:] @ Wq1[1024:1027,:] + bq1)
__global__ __launch_bounds__(256)
void point_layer1(const float* __restrict__ G0, const float* __restrict__ PP,
                  const float* __restrict__ Wq1, const float* __restrict__ bq1,
                  unsigned short* __restrict__ G1) {
  int r = blockIdx.x;          // 0..65535
  int b = r >> 4, p = r & 15;
  int j = threadIdx.x;         // 0..255
  float p0 = PP[b * 48 + p * 3 + 0];
  float p1 = PP[b * 48 + p * 3 + 1];
  float p2 = PP[b * 48 + p * 3 + 2];
  float v = G0[(size_t)b * 256 + j] + bq1[j];
  v = fmaf(p0, Wq1[(size_t)1024 * 256 + j], v);
  v = fmaf(p1, Wq1[(size_t)1025 * 256 + j], v);
  v = fmaf(p2, Wq1[(size_t)1026 * 256 + j], v);
  v = fmaxf(v, 0.0f);
  G1[(size_t)r * 256 + j] = f2bf(v);
}

// masks[b,p,j] = (j < floor(sigmoid(pp[b,p,0]) * 256)) ? 1 : 0
__global__ __launch_bounds__(256)
void mask_kernel(const float* __restrict__ PP, float* __restrict__ out) {
  int r = blockIdx.x;
  int b = r >> 4, p = r & 15;
  int j = threadIdx.x;
  float p0 = PP[b * 48 + p * 3 + 0];
  float n  = floorf(256.0f / (1.0f + __expf(-p0)));
  out[(size_t)r * 256 + j] = ((float)j < n) ? 1.0f : 0.0f;
}

extern "C" void kernel_launch(void* const* d_in, const int* in_sizes, int n_in,
                              void* d_out, int out_size, void* d_ws, size_t ws_size,
                              hipStream_t stream) {
  const float* latent = (const float*)d_in[0];
  const float* Wp1 = (const float*)d_in[1];  const float* bp1 = (const float*)d_in[2];
  const float* Wp2 = (const float*)d_in[3];  const float* bp2 = (const float*)d_in[4];
  const float* Wp3 = (const float*)d_in[5];  const float* bp3 = (const float*)d_in[6];
  const float* Wq1 = (const float*)d_in[7];  const float* bq1 = (const float*)d_in[8];
  const float* Wq2 = (const float*)d_in[9];  const float* bq2 = (const float*)d_in[10];
  const float* Wq3 = (const float*)d_in[11]; const float* bq3 = (const float*)d_in[12];

  char* ws = (char*)d_ws;
  unsigned short* latb = (unsigned short*)(ws + OFF_LATB);
  unsigned short* wp1t = (unsigned short*)(ws + OFF_WP1T);
  unsigned short* wp2t = (unsigned short*)(ws + OFF_WP2T);
  unsigned short* wp3t = (unsigned short*)(ws + OFF_WP3T);
  unsigned short* wq1t = (unsigned short*)(ws + OFF_WQ1T);
  unsigned short* wq2t = (unsigned short*)(ws + OFF_WQ2T);
  unsigned short* wq3t = (unsigned short*)(ws + OFF_WQ3T);
  unsigned short* h1   = (unsigned short*)(ws + OFF_H1);
  unsigned short* h2   = (unsigned short*)(ws + OFF_H2);
  float*          ppf  = (float*)(ws + OFF_PP);
  float*          g0f  = (float*)(ws + OFF_G0);
  unsigned short* g1   = (unsigned short*)(ws + OFF_G1);
  unsigned short* g2   = (unsigned short*)(ws + OFF_G2);

  float* out_pts   = (float*)d_out;                        // 4096*16*256*2
  float* out_masks = out_pts + (size_t)NROWS * NPOINTS * 2;

  // activation -> bf16; weights -> transposed (N-major, padded) bf16
  conv_f32_bf16<<<2048, 256, 0, stream>>>(latent, latb, BATCH * LATENT_DIM);
  transpose_bf16<<<2048, 256, 0, stream>>>(Wp1, wp1t, 1024, 512, 512);
  transpose_bf16<<<1024, 256, 0, stream>>>(Wp2, wp2t, 512, 512, 512);
  transpose_bf16<<<256,  256, 0, stream>>>(Wp3, wp3t, 512, 48, 128);
  transpose_bf16<<<1024, 256, 0, stream>>>(Wq1, wq1t, 1024, 256, 256); // rows 0..1023
  transpose_bf16<<<256,  256, 0, stream>>>(Wq2, wq2t, 256, 256, 256);
  transpose_bf16<<<512,  256, 0, stream>>>(Wq3, wq3t, 256, 512, 512);

  dim3 blk(256);
  // path decoder
  gemm_bf16_wmma<<<dim3(4, 32), blk, 0, stream>>>(latb, wp1t, bp1, nullptr, h1,
                                                  BATCH, 512, 1024, 1);
  gemm_bf16_wmma<<<dim3(4, 32), blk, 0, stream>>>(h1, wp2t, bp2, nullptr, h2,
                                                  BATCH, 512, 512, 1);
  gemm_bf16_wmma<<<dim3(1, 32), blk, 0, stream>>>(h2, wp3t, bp3, ppf, nullptr,
                                                  BATCH, 48, 512, 0);
  // shared part of point-decoder layer 1: latent @ Wq1[0:1024]
  gemm_bf16_wmma<<<dim3(2, 32), blk, 0, stream>>>(latb, wq1t, nullptr, g0f, nullptr,
                                                  BATCH, 256, 1024, 0);
  // per-path rank-3 correction + bias + relu
  point_layer1<<<NROWS, 256, 0, stream>>>(g0f, ppf, Wq1, bq1, g1);
  // point decoder layers 2,3
  gemm_bf16_wmma<<<dim3(2, 512), blk, 0, stream>>>(g1, wq2t, bq2, nullptr, g2,
                                                   NROWS, 256, 256, 1);
  gemm_bf16_wmma<<<dim3(4, 512), blk, 0, stream>>>(g2, wq3t, bq3, out_pts, nullptr,
                                                   NROWS, 512, 256, 2);
  // masks
  mask_kernel<<<NROWS, 256, 0, stream>>>(ppf, out_masks);
}
